// NodeEncoder_76244259438650
// MI455X (gfx1250) — matrix operand order repaired and checked by reading, compiled, not run on hardware
//
#include <hip/hip_runtime.h>
#include <hip/hip_bf16.h>

#define N_NODES   50000
#define N_EDGES   500000
#define EMB_DIM   288
#define NBASIS    32
#define NHID      32
#define WAVES     8
#define EPB       (WAVES * 16)   // edges per block

typedef __attribute__((ext_vector_type(16))) _Float16 v16h;
typedef __attribute__((ext_vector_type(8)))  _Float16 v8h;
typedef __attribute__((ext_vector_type(8)))  float    v8f;

__device__ __forceinline__ v16h cat8(v8h a, v8h b) {
    return __builtin_shufflevector(a, b, 0, 1, 2, 3, 4, 5, 6, 7,
                                         8, 9, 10, 11, 12, 13, 14, 15);
}

__global__ __launch_bounds__(256) void zero_out_kernel(float4* out, int n4) {
    int i = blockIdx.x * blockDim.x + threadIdx.x;
    if (i < n4) out[i] = make_float4(0.f, 0.f, 0.f, 0.f);
}

// atom[n, j] = z[n, 0:4] @ atom_W[0:4, j] + atom_b[j]
__global__ __launch_bounds__(256) void atom_kernel(const float* __restrict__ z,
                                                   const float* __restrict__ W,
                                                   const float* __restrict__ b,
                                                   float* __restrict__ atomBuf) {
    int i = blockIdx.x * blockDim.x + threadIdx.x;
    if (i >= N_NODES * EMB_DIM) return;
    int n = i / EMB_DIM, j = i - n * EMB_DIM;
    float acc = b[j];
    #pragma unroll
    for (int t = 0; t < 4; ++t) acc += z[n * 4 + t] * W[t * EMB_DIM + j];
    atomBuf[i] = acc;
}

__global__ __launch_bounds__(256) void edge_kernel(const float* __restrict__ pos,
                                                   const int*   __restrict__ ei,
                                                   const float* __restrict__ w0,
                                                   const float* __restrict__ w1,
                                                   const float* __restrict__ w2,
                                                   const float* __restrict__ mlpW1,
                                                   const float* __restrict__ mlpW2,
                                                   const float* __restrict__ atomBuf,
                                                   float* __restrict__ out) {
    // Transposed weight layouts: [column][K] so each lane's 16 K-values are
    // contiguous -> ds_load_b128 pairs instead of 16x ds_load_u16.
    __shared__ __align__(16) _Float16 sW1T[NHID][NBASIS];     //  2 KB
    __shared__ __align__(16) _Float16 sW2T[EMB_DIM][NHID];    // 18 KB
    __shared__ float    sw[96];                               // w0|w1|w2
    __shared__ __align__(16) _Float16 sRB[WAVES][16][NBASIS]; //  8 KB
    __shared__ __align__(16) _Float16 sH [WAVES][16][NHID];   //  8 KB
    __shared__ float    sY [WAVES][16][9];
    __shared__ float    sR [WAVES][16];
    __shared__ int      sSrc[WAVES][16];
    __shared__ int      sDst[WAVES][16];

    const int tid  = threadIdx.x;
    const int wave = tid >> 5;
    const int lane = tid & 31;
    const int hf   = lane >> 4;      // half-wave: 0 | 1
    const int lm   = lane & 15;

    const float rs32 = 0.17677669529663687f;     // 1/sqrt(32)
    for (int i = tid; i < NBASIS * NHID; i += 256) {
        int k = i >> 5, n = i & 31;              // mlpW1[k][n]
        sW1T[n][k] = (_Float16)(mlpW1[i] * rs32);
    }
    for (int i = tid; i < NHID * EMB_DIM; i += 256) {
        int k = i / EMB_DIM, j = i - k * EMB_DIM; // mlpW2[k][j]
        sW2T[j][k] = (_Float16)(mlpW2[i] * rs32);
    }
    for (int i = tid; i < 96; i += 256)
        sw[i] = (i < 32) ? w0[i] : (i < 64) ? w1[i - 32] : w2[i - 64];
    __syncthreads();

    const int base = blockIdx.x * EPB + wave * 16;

    // ---- geometry: lanes 0-15, one edge each ----
    if (lane < 16) {
        int  e     = base + lane;
        bool valid = (e < N_EDGES);
        int  ec    = valid ? e : 0;
        int  s = ei[ec], d = ei[N_EDGES + ec];
        sSrc[wave][lane] = s;
        sDst[wave][lane] = d;
        float vx = pos[3 * s + 0] - pos[3 * d + 0];
        float vy = pos[3 * s + 1] - pos[3 * d + 1];
        float vz = pos[3 * s + 2] - pos[3 * d + 2];
        float r = sqrtf(vx * vx + vy * vy + vz * vz);
        if (!valid) r = 0.0f;
        sR[wave][lane] = r;
        float inv = __builtin_amdgcn_rcpf(fmaxf(r, 1e-12f));
        float ux = vx * inv, uy = vy * inv, uz = vz * inv;
        const float s3  = 1.7320508075688772f;
        const float s5  = 2.2360679774997896f;
        const float s15 = 3.8729833462074170f;
        float Y[9];
        Y[0] = 1.0f;
        Y[1] = s3 * ux;  Y[2] = s3 * uy;  Y[3] = s3 * uz;
        Y[4] = s15 * ux * uy;
        Y[5] = s15 * uy * uz;
        Y[6] = 0.5f * s5 * (3.0f * uz * uz - 1.0f);
        Y[7] = s15 * ux * uz;
        Y[8] = 0.5f * s15 * (ux * ux - uy * uy);
        #pragma unroll
        for (int i = 0; i < 9; ++i) sY[wave][lane][i] = valid ? Y[i] : 0.0f;
    }
    __syncthreads();

    // ---- bessel basis: 2 lanes per edge, 16 freqs each; sqrt(2/c)=1 for c=2 ----
    {
        float r    = sR[wave][lm];
        float mask = (r > 0.0f && r < 2.0f) ? 1.0f : 0.0f;
        float coef = mask * __builtin_amdgcn_rcpf(fmaxf(r, 1e-12f));
        const float pio2 = 1.5707963267948966f;   // pi/c, c=2
        #pragma unroll
        for (int k = 0; k < 16; ++k) {
            int kk = hf * 16 + k;
            sRB[wave][lm][kk] = (_Float16)(coef * __sinf((float)(kk + 1) * pio2 * r));
        }
    }
    __syncthreads();

    // ---- A fragment, rb tile (16x32 f16): two contiguous 8-half chunks ----
    v16h aR = cat8(*(const v8h*)&sRB[wave][lm][hf * 8],
                   *(const v8h*)&sRB[wave][lm][16 + hf * 8]);

    // ---- GEMM1: H = SILU_NORM * silu(RB @ W1s), two 16-col tiles ----
    #pragma unroll
    for (int nt = 0; nt < 2; ++nt) {
        const _Float16* col = &sW1T[nt * 16 + lm][hf * 8];
        v16h b = cat8(*(const v8h*)col, *(const v8h*)(col + 16));
        v8f c = {};
        c = __builtin_amdgcn_wmma_f32_16x16x32_f16(false, aR, false, b, (short)0, c, false, false);
        #pragma unroll
        for (int k = 0; k < 8; ++k) {
            float xh = c[k];
            float hv = 1.679177f * xh * __builtin_amdgcn_rcpf(1.0f + __expf(-xh));
            sH[wave][k + 8 * hf][nt * 16 + lm] = (_Float16)hv;
        }
    }
    __syncthreads();

    // ---- A fragment, hidden tile (16x32 f16) ----
    v16h aH = cat8(*(const v8h*)&sH[wave][lm][hf * 8],
                   *(const v8h*)&sH[wave][lm][16 + hf * 8]);

    // Per-row metadata for this lane's 8 output rows (M = k + 8*hf)
    int   rSrc[8], rDst[8];
    bool  rOk[8];
    #pragma unroll
    for (int k = 0; k < 8; ++k) {
        int M   = k + 8 * hf;
        rOk[k]  = (base + M) < N_EDGES;
        rSrc[k] = sSrc[wave][M];
        rDst[k] = sDst[wave][M];
    }

    // ---- GEMM2: radial = H @ W2s, 18 tiles; fuse sph*atom, atomic scatter ----
    for (int t = 0; t < 18; ++t) {
        const int j = t * 16 + lm;               // output column owned by this lane
        const _Float16* col = &sW2T[j][hf * 8];
        v16h b = cat8(*(const v8h*)col, *(const v8h*)(col + 16));

        // hoist the 8 atom gathers so they overlap the WMMA
        float av[8];
        #pragma unroll
        for (int k = 0; k < 8; ++k) av[k] = atomBuf[rSrc[k] * EMB_DIM + j];

        v8f c = {};
        c = __builtin_amdgcn_wmma_f32_16x16x32_f16(false, aH, false, b, (short)0, c, false, false);

        float wf; int yidx;
        if (j < 32)       { wf = sw[j];                      yidx = 0; }
        else if (j < 128) { int jj = j - 32;  wf = sw[32 + jj / 3]; yidx = 1 + jj % 3; }
        else              { int jj = j - 128; wf = sw[64 + jj / 5]; yidx = 4 + jj % 5; }

        #pragma unroll
        for (int k = 0; k < 8; ++k) {
            int M = k + 8 * hf;
            if (rOk[k]) {
                float val = c[k] * wf * sY[wave][M][yidx] * av[k];
                atomicAdd(&out[rDst[k] * EMB_DIM + j], val);
            }
        }
    }
}

extern "C" void kernel_launch(void* const* d_in, const int* in_sizes, int n_in,
                              void* d_out, int out_size, void* d_ws, size_t ws_size,
                              hipStream_t stream) {
    const float* pos    = (const float*)d_in[0];
    const float* z      = (const float*)d_in[1];
    const int*   ei     = (const int*)  d_in[2];
    const float* w0     = (const float*)d_in[3];
    const float* w1     = (const float*)d_in[4];
    const float* w2     = (const float*)d_in[5];
    const float* atom_W = (const float*)d_in[6];
    const float* atom_b = (const float*)d_in[7];
    const float* mlpW1  = (const float*)d_in[8];
    const float* mlpW2  = (const float*)d_in[9];
    float*       out    = (float*)d_out;
    float*       atomBuf = (float*)d_ws;          // N_NODES*EMB_DIM floats = 57.6 MB

    const int nOut = N_NODES * EMB_DIM;
    zero_out_kernel<<<(nOut / 4 + 255) / 256, 256, 0, stream>>>((float4*)out, nOut / 4);
    atom_kernel<<<(nOut + 255) / 256, 256, 0, stream>>>(z, atom_W, atom_b, atomBuf);
    edge_kernel<<<(N_EDGES + EPB - 1) / EPB, 256, 0, stream>>>(
        pos, ei, w0, w1, w2, mlpW1, mlpW2, atomBuf, out);
}
